// Convblock_14130442404231
// MI455X (gfx1250) — matrix operand order repaired and checked by reading, not compiled, not run on hardware
//
#include <hip/hip_runtime.h>
#include <math.h>

typedef __attribute__((ext_vector_type(16))) _Float16 v16h;
typedef __attribute__((ext_vector_type(8)))  _Float16 v8h;
typedef __attribute__((ext_vector_type(8)))  float    v8f;
typedef __attribute__((ext_vector_type(4)))  unsigned int u32x4;
typedef __attribute__((ext_vector_type(8)))  int          i32x8;
typedef __attribute__((ext_vector_type(4)))  int          i32x4;

namespace {
constexpr int kB = 4, kT = 65536, kC = 64, kK = 7;
constexpr int ROWS  = 128;              // time rows per workgroup
constexpr int HALO  = 3;                // K//2
constexpr int LROWS = ROWS + 2 * HALO;  // 134
constexpr int XS = 68;                  // fp32 LDS row stride (floats) -> conflict-free
constexpr int HS = 72;                  // fp16 LDS row stride (halves) -> 16B-aligned b128 groups
constexpr int GATE_HALVES = kK * 4 * 2 * 32 * 16;  // 28672 packed gate-weight halves
constexpr int CONV_HALVES = 4 * 2 * 32 * 16;       // 4096 packed conv-weight halves
}

#if __has_builtin(__builtin_amdgcn_tanhf)
#define DEV_TANHF(x) __builtin_amdgcn_tanhf(x)
#else
#define DEV_TANHF(x) tanhf(x)
#endif

// ---------------------------------------------------------------------------
// Pack fp32 weights into exact WMMA B-fragment layout (f16) in workspace.
// B fragment (16x16x32 f16): lane l -> N = l&15, half h -> K = 8*(l>>4) + (h&7) + 16*(h>>3).
// Gate frag index: (((k*4 + ct)*2 + frag)*32 + lane)*16 + h,
//   K-dim = c_in = 8*(lane>>4) + (h&7) + 16*(h>>3) + 32*frag, N = d = 16*ct + (lane&15),
//   value = weights[d, c, k]   (shape (C_out, C_in, K), row-major)
// Conv frags appended at GATE_HALVES, value = conv_w[c, d].
// ---------------------------------------------------------------------------
__global__ void pack_weights_kernel(const float* __restrict__ w,
                                    const float* __restrict__ cw,
                                    _Float16* __restrict__ pk) {
  const int tid = blockIdx.x * blockDim.x + threadIdx.x;
  if (tid < GATE_HALVES) {
    const int h    = tid & 15;
    const int lane = (tid >> 4) & 31;
    const int frag = (tid >> 9) & 1;
    const int ct   = (tid >> 10) & 3;
    const int k    = tid >> 12;
    const int d = 16 * ct + (lane & 15);
    const int c = (h & 7) + 16 * (h >> 3) + 8 * (lane >> 4) + 32 * frag;
    pk[tid] = (_Float16)w[(d * kC + c) * kK + k];
  } else if (tid < GATE_HALVES + CONV_HALVES) {
    const int e    = tid - GATE_HALVES;
    const int h    = e & 15;
    const int lane = (e >> 4) & 31;
    const int frag = (e >> 9) & 1;
    const int ct   = (e >> 10) & 3;
    const int c = (h & 7) + 16 * (h >> 3) + 8 * (lane >> 4) + 32 * frag;
    const int d = 16 * ct + (lane & 15);
    pk[tid] = (_Float16)cw[c * kC + d];
  }
}

// ---------------------------------------------------------------------------
// Fused main kernel: TDM-staged x tile -> gate GEMM (WMMA f16) + tanh +
// windowed gather + skips + LayerNorm (shuffle reduction) + 1x1 conv (WMMA)
// + PReLU + outer skip. One block = 128 rows of one batch; 8 waves x 16 rows.
// ---------------------------------------------------------------------------
__global__ __launch_bounds__(256, 1)
void convblock_main(const float* __restrict__ x,
                    const float* __restrict__ ln_s,
                    const float* __restrict__ ln_b,
                    const float* __restrict__ cbias,
                    const float* __restrict__ slope_p,
                    const _Float16* __restrict__ pk,
                    float* __restrict__ out) {
  __shared__ float    xf[LROWS][XS];   // fp32 x tile incl. halo (stride 68)
  __shared__ _Float16 y16[ROWS][HS];   // fp16 staging for LN'd y (conv A frags)

  const int tid = threadIdx.x;
  const int b   = blockIdx.x / (kT / ROWS);
  const int t0  = (blockIdx.x % (kT / ROWS)) * ROWS;

  // Warm L0/L2 with the packed weights (global_prefetch_b8 path).
  __builtin_prefetch(pk + (tid << 7), 0, 0);

  // -------- Phase 1: stage x tile (134 rows x 64 ch fp32) into LDS --------
#if __has_builtin(__builtin_amdgcn_tensor_load_to_lds)
  if (t0 > 0) {
    // Tensor Data Mover: one 2D descriptor moves the whole tile.
    //  - rows beyond the tensor end (last block) zero-fill via OOB handling
    //  - LDS pad: after every 64 DW (one 256B row) insert 4 DW -> stride 68
    if (tid == 0) {
      const unsigned lds_off = (unsigned)(uintptr_t)&xf[0][0];
      const unsigned long long ga =
          (unsigned long long)(uintptr_t)x +
          ((unsigned long long)(b * kT + t0 - HALO) * kC * 4u);
      const unsigned rows_avail = (unsigned)(kT + HALO - t0);  // >= 131

      u32x4 g0;
      g0[0] = 1u;                               // count=1, user-mode load
      g0[1] = lds_off;                          // lds_addr (bytes)
      g0[2] = (unsigned)ga;                     // global_addr[31:0]
      g0[3] = (unsigned)((ga >> 32) & 0x1FFFFFFu) | (2u << 30);  // addr[56:32] | type=2

      i32x8 g1;
      g1[0] = (int)((2u << 16)        // data_size = 4B
                  | (1u << 20)        // pad_enable
                  | (5u << 22)        // pad_interval code 5 -> every 64 DW
                  | (3u << 25));      // pad_amount  code 3 -> 4 DW
      g1[1] = (int)(64u << 16);                      // tensor_dim0[15:0] = 64
      g1[2] = (int)(rows_avail << 16);               // dim0 hi=0 | tensor_dim1 lo
      g1[3] = (int)((rows_avail >> 16) | (64u << 16)); // tensor_dim1 hi | tile_dim0=64
      g1[4] = (int)(unsigned)LROWS;                  // tile_dim1=134, tile_dim2=0
      g1[5] = 64;                                    // tensor_dim0_stride lo
      g1[6] = 0;                                     // stride0 hi | stride1 lo
      g1[7] = 0;
      const i32x4 gz4 = {0, 0, 0, 0};
      const i32x8 gz8 = {0, 0, 0, 0, 0, 0, 0, 0};
      __builtin_amdgcn_tensor_load_to_lds(g0, g1, gz4, gz4, gz8, 0);
    }
#if __has_builtin(__builtin_amdgcn_s_wait_tensorcnt)
    __builtin_amdgcn_s_wait_tensorcnt((short)0);
#else
    asm volatile("s_wait_tensorcnt 0x0" ::: "memory");
#endif
  } else
#endif
  {
    // Scalar fallback: first block of each batch (negative halo) or no TDM.
    for (int j = tid; j < LROWS * (kC / 4); j += 256) {
      const int i  = j >> 4;
      const int c4 = (j & 15) << 2;
      const int tg = t0 - HALO + i;
      float4 v = make_float4(0.f, 0.f, 0.f, 0.f);
      if (tg >= 0 && tg < kT)
        v = *(const float4*)(x + ((size_t)(b * kT + tg) * kC + c4));
      *(float4*)&xf[i][c4] = v;
    }
  }
  __syncthreads();

  const int w    = tid >> 5;
  const int l    = tid & 31;
  const int ln16 = l & 15;
  const int hi   = l >> 4;
  const int rb   = w << 4;   // first local row of this wave's M tile

  union AF { v16h v; v8h h[2]; };
  auto cvt8 = [](float4 a, float4 c) -> v8h {
    v8h r;
    r[0] = (_Float16)a.x; r[1] = (_Float16)a.y;
    r[2] = (_Float16)a.z; r[3] = (_Float16)a.w;
    r[4] = (_Float16)c.x; r[5] = (_Float16)c.y;
    r[6] = (_Float16)c.z; r[7] = (_Float16)c.w;
    return r;
  };

  // A fragments of x (f16 from fp32 LDS): lane = M row; halves 0-7 ->
  // K = 8*hi + 0..7, halves 8-15 -> K = 16 + 8*hi + 0..7 (second frag +32).
  AF a0, a1;
  {
    const float* base = &xf[rb + ln16 + HALO][8 * hi];
    a0.h[0] = cvt8(*(const float4*)(base +  0), *(const float4*)(base +  4));
    a0.h[1] = cvt8(*(const float4*)(base + 16), *(const float4*)(base + 20));
    a1.h[0] = cvt8(*(const float4*)(base + 32), *(const float4*)(base + 36));
    a1.h[1] = cvt8(*(const float4*)(base + 48), *(const float4*)(base + 52));
  }

  const v8f zero8 = {0.f, 0.f, 0.f, 0.f, 0.f, 0.f, 0.f, 0.f};
  v8f yacc[4] = {zero8, zero8, zero8, zero8};

  // -------- Phase 2: gate GEMM + tanh + windowed gather-multiply --------
  for (int k = 0; k < kK; ++k) {
    #pragma unroll
    for (int ct = 0; ct < 4; ++ct) {
      const _Float16* bp = pk + (((k * 4 + ct) * 2) * 32 + l) * 16;
      AF b0, b1;
      b0.h[0] = *(const v8h*)(bp);
      b0.h[1] = *(const v8h*)(bp + 8);
      b1.h[0] = *(const v8h*)(bp + 512);
      b1.h[1] = *(const v8h*)(bp + 520);
      v8f acc = zero8;
      acc = __builtin_amdgcn_wmma_f32_16x16x32_f16(false, a0.v, false, b0.v,
                                                   (short)0, acc, false, false);
      acc = __builtin_amdgcn_wmma_f32_16x16x32_f16(false, a1.v, false, b1.v,
                                                   (short)0, acc, false, false);
      #pragma unroll
      for (int r = 0; r < 8; ++r) {
        // D layout: row = rb + r + 8*hi, col = 16*ct + ln16.
        // xu = x[t + k - 3, c]; local xf row = (rb + r + 8*hi) + k (halo folds in).
        const float g  = DEV_TANHF(acc[r]);
        const float xu = xf[rb + r + 8 * hi + k][ct * 16 + ln16];
        yacc[ct][r] += g * xu;
      }
    }
  }

  // Per-lane channel-indexed parameters (c = 16*ct + ln16).
  float lnsc[4], lnbi[4], cb4[4];
  #pragma unroll
  for (int ct = 0; ct < 4; ++ct) {
    lnsc[ct] = ln_s[ct * 16 + ln16];
    lnbi[ct] = ln_b[ct * 16 + ln16];
    cb4[ct]  = cbias[ct * 16 + ln16];
  }
  const float slope = slope_p[0];

  // -------- Phase 3: inner skip + LayerNorm (shuffle reduction over C=64) ----
  #pragma unroll
  for (int r = 0; r < 8; ++r) {
    #pragma unroll
    for (int ct = 0; ct < 4; ++ct)
      yacc[ct][r] += xf[rb + r + 8 * hi + HALO][ct * 16 + ln16];  // y = x + y
    float s = yacc[0][r] + yacc[1][r] + yacc[2][r] + yacc[3][r];
    float q = yacc[0][r] * yacc[0][r] + yacc[1][r] * yacc[1][r] +
              yacc[2][r] * yacc[2][r] + yacc[3][r] * yacc[3][r];
    #pragma unroll
    for (int m = 1; m < 16; m <<= 1) {   // reduce within each 16-lane N group
      s += __shfl_xor(s, m, 32);
      q += __shfl_xor(q, m, 32);
    }
    const float mean = s * (1.f / 64.f);
    const float var  = q * (1.f / 64.f) - mean * mean;
    const float rstd = rsqrtf(var + 1e-6f);
    #pragma unroll
    for (int ct = 0; ct < 4; ++ct) {
      const float v = (yacc[ct][r] - mean) * rstd * lnsc[ct] + lnbi[ct];
      yacc[ct][r] = v;
      // Re-layout LN'd y into fp16 LDS (this wave's private rows) for conv A.
      y16[rb + r + 8 * hi][ct * 16 + ln16] = (_Float16)v;
    }
  }

  // A fragments of normalized y.
  AF ay0, ay1;
  {
    const int rr = rb + ln16;
    ay0.h[0] = *(const v8h*)&y16[rr][ 8 * hi];
    ay0.h[1] = *(const v8h*)&y16[rr][16 + 8 * hi];
    ay1.h[0] = *(const v8h*)&y16[rr][32 + 8 * hi];
    ay1.h[1] = *(const v8h*)&y16[rr][48 + 8 * hi];
  }

  // -------- Phase 4: 1x1 conv (WMMA) + bias + PReLU + outer skip, store ------
  float* outw = out + (size_t)(b * kT + t0 + rb + 8 * hi) * kC + ln16;
  #pragma unroll
  for (int ct = 0; ct < 4; ++ct) {
    const _Float16* bp = pk + GATE_HALVES + ((ct * 2) * 32 + l) * 16;
    AF b0, b1;
    b0.h[0] = *(const v8h*)(bp);
    b0.h[1] = *(const v8h*)(bp + 8);
    b1.h[0] = *(const v8h*)(bp + 512);
    b1.h[1] = *(const v8h*)(bp + 520);
    v8f zc = zero8;
    zc = __builtin_amdgcn_wmma_f32_16x16x32_f16(false, ay0.v, false, b0.v,
                                                (short)0, zc, false, false);
    zc = __builtin_amdgcn_wmma_f32_16x16x32_f16(false, ay1.v, false, b1.v,
                                                (short)0, zc, false, false);
    #pragma unroll
    for (int r = 0; r < 8; ++r) {
      float z = zc[r] + cb4[ct];
      z = (z >= 0.f) ? z : slope * z;
      outw[(size_t)r * kC + ct * 16] = yacc[ct][r] + z;
    }
  }
}

extern "C" void kernel_launch(void* const* d_in, const int* in_sizes, int n_in,
                              void* d_out, int out_size, void* d_ws, size_t ws_size,
                              hipStream_t stream) {
  (void)in_sizes; (void)n_in; (void)out_size; (void)ws_size;
  const float* x      = (const float*)d_in[0];
  const float* wts    = (const float*)d_in[1];
  const float* ln_s   = (const float*)d_in[2];
  const float* ln_b   = (const float*)d_in[3];
  const float* conv_w = (const float*)d_in[4];
  const float* conv_b = (const float*)d_in[5];
  const float* slope  = (const float*)d_in[6];
  float* out = (float*)d_out;
  _Float16* pk = (_Float16*)d_ws;  // 65536 B of packed f16 weight fragments

  pack_weights_kernel<<<(GATE_HALVES + CONV_HALVES) / 256, 256, 0, stream>>>(
      wts, conv_w, pk);
  convblock_main<<<kB * (kT / ROWS), 256, 0, stream>>>(
      x, ln_s, ln_b, conv_b, slope, pk, out);
}
